// NittaConv2d_43224550867497
// MI455X (gfx1250) — matrix operand order repaired and checked by reading, not compiled, not run on hardware
//
#include <hip/hip_runtime.h>

typedef __attribute__((ext_vector_type(16))) _Float16 v16h;
typedef __attribute__((ext_vector_type(8)))  _Float16 v8h;
typedef __attribute__((ext_vector_type(8)))  float    v8f;

#define CI   128
#define CO   256
#define HH_  112
#define WW_  112
#define NB   32
#define HW   (HH_ * WW_)        // 12544
#define WP   114                // padded spatial (1-px halo)
#define BM   128                // co tile
#define BN   64                 // pixel tile (row-aligned)
#define BK   64                 // k chunk (2 WMMA k-steps)
#define ASTR 72                 // LDS row stride (halfs): 144B, bank-friendly
#define BSTR 72
#define AHALFS (BM * ASTR)      // 9216
#define BHALFS (BN * BSTR)      // 4608
#define BUFH   (AHALFS + BHALFS)// 13824 halfs / buffer (27KB); x2 = 54KB

// ---------------- CDNA5 async copy helpers ----------------
__device__ __forceinline__ void async_ld_b128(unsigned lds_off, const void* gptr) {
    asm volatile("global_load_async_to_lds_b128 %0, %1, off"
                 :: "v"(lds_off), "v"(gptr) : "memory");
}
__device__ __forceinline__ void wait_async_6() {
    asm volatile("s_wait_asynccnt 6" ::: "memory");
}
__device__ __forceinline__ void wait_async_0() {
    asm volatile("s_wait_asynccnt 0" ::: "memory");
}

// ---- prepass 1: fp32 weight [co][ci][fy][fx] -> f16 [t][co][ci] ----
__global__ void __launch_bounds__(256)
prepack_weights(const float* __restrict__ w, _Float16* __restrict__ wh) {
    int idx = blockIdx.x * 256 + threadIdx.x;   // 294912 total
    int t   = idx >> 15;
    int r   = idx & 32767;
    int co  = r >> 7;
    int ci  = r & 127;
    wh[idx] = (_Float16)w[co * (CI * 9) + ci * 9 + t];
}

// ---- prepass 2: fp32 NCHW -> f16 zero-padded NHWC [n][yy:114][xx:114][ci] ----
__global__ void __launch_bounds__(256)
prepack_input(const float* __restrict__ img, _Float16* __restrict__ imgp) {
    const int tid = threadIdx.x;
    const int b   = blockIdx.x;              // 32*114 blocks: one padded row each
    const int n   = b / WP;
    const int yy  = b % WP;
    _Float16* dst = imgp + (size_t)(n * WP + yy) * WP * CI;
    const int NV  = (WP * CI) / 8;           // 1824 v8h per row
    if (yy == 0 || yy == WP - 1) {
        v8h z = {};
        for (int i = tid; i < NV; i += 256) ((v8h*)dst)[i] = z;
        return;
    }
    __shared__ __align__(16) _Float16 lt[WP * 136];
    const int y = yy - 1;
    if (tid < CI) {                          // halo columns
        lt[0 * 136 + tid]   = (_Float16)0.f;
        lt[113 * 136 + tid] = (_Float16)0.f;
    }
    {
        const int x   = tid & 127;           // coalesced along x
        const int ci0 = tid >> 7;            // 0..1
        if (x < WW_) {
            for (int ci = ci0; ci < CI; ci += 2)
                lt[(x + 1) * 136 + ci] =
                    (_Float16)img[((size_t)(n * CI + ci) * HH_ + y) * WW_ + x];
        }
    }
    __syncthreads();
    for (int e = tid; e < NV; e += 256) {
        int xx = e >> 4;                     // 16 v8h per xx
        int q  = e & 15;
        ((v8h*)dst)[e] = *(const v8h*)&lt[xx * 136 + q * 8];
    }
}

// ---- main conv: async double-buffered implicit GEMM, f16 WMMA ----
__global__ void __launch_bounds__(256)
conv_wmma_async(const _Float16* __restrict__ wh, const _Float16* __restrict__ imgp,
                const float* __restrict__ bias, float* __restrict__ out) {
    __shared__ __align__(16) _Float16 lds[2 * BUFH];

    const int tid  = threadIdx.x;
    const int lane = tid & 31;
    const int wave = tid >> 5;
    const int wm   = wave & 3;      // co sub-block (4 x 32 = 128)
    const int wn   = wave >> 2;     // px sub-block (2 x 32 = 64)
    const int hh   = lane >> 4;
    const int l16  = lane & 15;

    const int tix = blockIdx.x;              // 32*112*2 = 7168
    const int co0 = blockIdx.y * BM;
    const int n   = tix / 224;
    const int rr  = tix % 224;
    const int y   = rr >> 1;
    const int x0  = (rr & 1) * 48;           // overlapping row-aligned tiles

    // per-thread async addressing (all 16B aligned)
    const int k8      = (tid & 7) * 8;       // half offset inside a b128
    const int row8    = tid >> 3;            // 0..31
    const size_t aG   = (size_t)(co0 + row8) * CI + k8;
    const unsigned aL = (unsigned)((row8 * ASTR + k8) * 2);
    const size_t bGl  = (size_t)(x0 + row8) * CI + k8;     // lane part of B addr
    const unsigned bL = (unsigned)(AHALFS * 2 + (row8 * BSTR + k8) * 2);

    auto issue = [&](int stage, int bufsel) {
        const int t  = stage >> 1, c = stage & 1;
        const int fy = t / 3,      fx = t % 3;
        const unsigned boff = (unsigned)(bufsel * (BUFH * 2));
        const _Float16* gA = wh + (size_t)t * CO * CI + (size_t)c * BK + aG;
#pragma unroll
        for (int i = 0; i < 4; ++i)
            async_ld_b128(aL + boff + i * (32 * ASTR * 2), gA + (size_t)i * 32 * CI);
        const _Float16* gB = imgp + ((size_t)(n * WP + y + fy) * WP + fx) * CI
                                  + (size_t)c * BK + bGl;
#pragma unroll
        for (int i = 0; i < 2; ++i)
            async_ld_b128(bL + boff + i * (32 * BSTR * 2), gB + (size_t)i * 32 * CI);
    };

    v8f acc[2][2] = {};

    issue(0, 0);
    for (int s = 0; s < 18; ++s) {           // t = s>>1 (9 taps), c = s&1 (2 k-chunks)
        __syncthreads();                     // everyone done computing stage s-1
        if (s + 1 < 18) issue(s + 1, (s + 1) & 1);
        if (s + 1 < 18) wait_async_6(); else wait_async_0();  // stage-s data landed
        __syncthreads();                     // ... in every wave
        const _Float16* Ab = lds + (s & 1) * BUFH;
        const _Float16* Bb = Ab + AHALFS;
#pragma unroll
        for (int ks = 0; ks < 2; ++ks) {
            v16h af[2], bf[2];
#pragma unroll
            for (int mt = 0; mt < 2; ++mt) {
                const _Float16* ar = Ab + (wm * 32 + mt * 16 + l16) * ASTR + ks * 32;
                union { v16h v; v8h h[2]; } ua;
                ua.h[0] = *(const v8h*)(ar + 8 * hh);
                ua.h[1] = *(const v8h*)(ar + 8 * hh + 16);
                af[mt] = ua.v;
            }
#pragma unroll
            for (int nt = 0; nt < 2; ++nt) {
                const _Float16* br = Bb + (wn * 32 + nt * 16 + l16) * BSTR + ks * 32;
                union { v16h v; v8h h[2]; } ub;
                ub.h[0] = *(const v8h*)(br + 8 * hh);
                ub.h[1] = *(const v8h*)(br + 8 * hh + 16);
                bf[nt] = ub.v;
            }
#pragma unroll
            for (int mt = 0; mt < 2; ++mt)
#pragma unroll
                for (int nt = 0; nt < 2; ++nt)
                    acc[mt][nt] = __builtin_amdgcn_wmma_f32_16x16x32_f16(
                        false, af[mt], false, bf[nt], (short)0, acc[mt][nt], false, false);
        }
    }

    // epilogue: lane holds N=l16 (x), VGPR r -> M = r + 8*hh (co)
#pragma unroll
    for (int mt = 0; mt < 2; ++mt) {
#pragma unroll
        for (int nt = 0; nt < 2; ++nt) {
            int xcol = x0 + wn * 32 + nt * 16 + l16;
            int cob  = co0 + wm * 32 + mt * 16 + 8 * hh;
#pragma unroll
            for (int r = 0; r < 8; ++r) {
                int co = cob + r;
                out[((size_t)n * CO + co) * HW + y * WW_ + xcol] =
                    acc[mt][nt][r] + bias[co];
            }
        }
    }
}

// ---- fallback (no/small workspace): round-1 kernel, in-loop convert ----
__global__ void __launch_bounds__(256)
conv_wmma_fb(const float* __restrict__ img, const float* __restrict__ wgt,
             const float* __restrict__ bias, float* __restrict__ out) {
    __shared__ __align__(16) _Float16 Ah[BM * 40];
    __shared__ __align__(16) _Float16 Bh[64 * 40];
    const int tid  = threadIdx.x;
    const int lane = tid & 31, wave = tid >> 5;
    const int wm = wave & 3, wn = wave >> 2, hh = lane >> 4, l16 = lane & 15;
    const int p0 = blockIdx.x * 64, co0 = blockIdx.y * BM;
    const int pxl = tid & 63, p = p0 + pxl;
    const int n = p / HW, rem = p % HW, y = rem / WW_, x = rem % WW_;
    const int kk0 = tid >> 6;
    v8f acc[2][2] = {};
    for (int t = 0; t < 9; ++t) {
        const int fy = t / 3, fx = t % 3;
        const int yy = y + fy - 1, xx = x + fx - 1;
        const bool valid = ((unsigned)yy < HH_) && ((unsigned)xx < WW_);
        const float* bsrc = img + ((long)(n * CI) * HH_ + yy) * WW_ + xx;
        for (int c = 0; c < 4; ++c) {
            __syncthreads();
            const float* s = bsrc + (long)(c * 32) * HW;
#pragma unroll
            for (int i = 0; i < 8; ++i) {
                int kk = kk0 + i * 4;
                Bh[pxl * 40 + kk] = (_Float16)(valid ? s[(long)kk * HW] : 0.0f);
            }
#pragma unroll
            for (int i = 0; i < 16; ++i) {
                int e = i * 256 + tid, com = e >> 5, kk = e & 31;
                Ah[com * 40 + kk] =
                    (_Float16)wgt[(co0 + com) * (CI * 9) + (c * 32 + kk) * 9 + t];
            }
            __syncthreads();
            v16h af[2], bf[2];
#pragma unroll
            for (int mt = 0; mt < 2; ++mt) {
                const _Float16* ar = &Ah[(wm * 32 + mt * 16 + l16) * 40];
                union { v16h v; v8h h[2]; } ua;
                ua.h[0] = *(const v8h*)(ar + 8 * hh);
                ua.h[1] = *(const v8h*)(ar + 8 * hh + 16);
                af[mt] = ua.v;
            }
#pragma unroll
            for (int nt = 0; nt < 2; ++nt) {
                const _Float16* br = &Bh[(wn * 32 + nt * 16 + l16) * 40];
                union { v16h v; v8h h[2]; } ub;
                ub.h[0] = *(const v8h*)(br + 8 * hh);
                ub.h[1] = *(const v8h*)(br + 8 * hh + 16);
                bf[nt] = ub.v;
            }
#pragma unroll
            for (int mt = 0; mt < 2; ++mt)
#pragma unroll
                for (int nt = 0; nt < 2; ++nt)
                    acc[mt][nt] = __builtin_amdgcn_wmma_f32_16x16x32_f16(
                        false, af[mt], false, bf[nt], (short)0, acc[mt][nt], false, false);
        }
    }
#pragma unroll
    for (int mt = 0; mt < 2; ++mt)
#pragma unroll
        for (int nt = 0; nt < 2; ++nt) {
            int px = p0 + wn * 32 + nt * 16 + l16;
            int n2 = px / HW, r2 = px % HW;
            long obase = (long)n2 * CO * HW + r2;
            int cob = co0 + wm * 32 + mt * 16 + 8 * hh;
#pragma unroll
            for (int r = 0; r < 8; ++r) {
                int co = cob + r;
                out[obase + (long)co * HW] = acc[mt][nt][r] + bias[co];
            }
        }
}

extern "C" void kernel_launch(void* const* d_in, const int* in_sizes, int n_in,
                              void* d_out, int out_size, void* d_ws, size_t ws_size,
                              hipStream_t stream) {
    const float* img  = (const float*)d_in[0];
    const float* wgt  = (const float*)d_in[1];
    const float* bias = (const float*)d_in[2];
    float* out = (float*)d_out;

    const size_t wbytes = (size_t)9 * CO * CI * 2u;           // 576 KB
    const size_t ibytes = (size_t)NB * WP * WP * CI * 2u;     // ~101.5 MiB
    dim3 block(256);

    if (d_ws && ws_size >= wbytes + ibytes) {
        _Float16* whp  = (_Float16*)d_ws;
        _Float16* ipad = (_Float16*)((char*)d_ws + wbytes);   // 16B-aligned offset
        prepack_weights<<<(9 * CO * CI) / 256, block, 0, stream>>>(wgt, whp);
        prepack_input<<<NB * WP, block, 0, stream>>>(img, ipad);
        conv_wmma_async<<<dim3(NB * HH_ * 2, CO / BM), block, 0, stream>>>(whp, ipad, bias, out);
    } else {
        conv_wmma_fb<<<dim3((NB * HW) / 64, CO / BM), block, 0, stream>>>(img, wgt, bias, out);
    }
}
